// VQCodebook_4277787427493
// MI455X (gfx1250) — matrix operand order repaired and checked by reading, not compiled, not run on hardware
//
#include <hip/hip_runtime.h>
#include <hip/hip_bf16.h>
#include <math.h>

// ---------------- problem constants ----------------
#define BSZ   32
#define WLEN  2048
#define NROWS (BSZ*WLEN)      // 65536
#define FD    256             // feature dim
#define SLOTS 1024
#define TS    128             // codebook slots per LDS tile
#define NT    (SLOTS/TS)      // 8 tiles
#define RPW   16              // rows per wave
#define NWAVES 8
#define RPB   (RPW*NWAVES)    // 128 rows per block

// LDS strides (elements), padded so 16B fragment loads hit 4-bank strides
#define CB_STRIDE  264        // bf16, [TS][CB_STRIDE]
#define CBT_STRIDE 136        // bf16, [FD][CBT_STRIDE]
#define ZA_STRIDE  264        // bf16, per wave [RPW][ZA_STRIDE]
#define TT_STRIDE  132        // f32,  per wave [RPW][TT_STRIDE]
#define WT_STRIDE  136        // bf16, per wave [RPW][WT_STRIDE]

typedef __attribute__((ext_vector_type(16))) __bf16 v16bf;
typedef __attribute__((ext_vector_type(8)))  __bf16 v8bf;
typedef __attribute__((ext_vector_type(8)))  float  v8f;

// byte offsets inside the shared block
#define CB_OFF   0
#define CBT_OFF  (CB_OFF  + TS*CB_STRIDE*2)                 // 67584
#define ZA_OFF   (CBT_OFF + FD*CBT_STRIDE*2)                // +69632
#define TT_OFF   (ZA_OFF  + NWAVES*RPW*ZA_STRIDE*2)         // +67584
#define WT_OFF   (TT_OFF  + NWAVES*RPW*TT_STRIDE*4)         // +67584
#define CSQ_OFF  (WT_OFF  + NWAVES*RPW*WT_STRIDE*2)         // +34816
#define ZSQ_OFF  (CSQ_OFF + TS*4)
#define SMEM_TOT (ZSQ_OFF + NWAVES*RPW*4)                   // 308224 bytes

static __device__ __forceinline__ v16bf pack16(v8bf lo, v8bf hi){
  v16bf r;
#pragma unroll
  for (int i=0;i<8;++i){ r[i]=lo[i]; r[i+8]=hi[i]; }
  return r;
}
// 16-bit A fragment (16x32): lane m holds K in {16*kh .. 16*kh+15} (contiguous)
static __device__ __forceinline__ v16bf fragA(const __bf16* rowbase, int k0, int kh){
  v8bf lo = *(const v8bf*)(rowbase + k0 + 16*kh);
  v8bf hi = *(const v8bf*)(rowbase + k0 + 16*kh + 8);
  return pack16(lo,hi);
}
// 16-bit B fragment (32x16): lane n holds K in {16*kh .. 16*kh+15} (contiguous)
static __device__ __forceinline__ v16bf fragB(const __bf16* colbase, int k0, int kh){
  const __bf16* p = colbase + k0 + 16*kh;
  return pack16(*(const v8bf*)p, *(const v8bf*)(p+8));
}
static __device__ __forceinline__ float bsum16(float x){
#pragma unroll
  for (int m=1;m<16;m<<=1) x += __shfl_xor(x, m);
  return x;
}
static __device__ __forceinline__ void bargmax16(float& v, int& i){
#pragma unroll
  for (int m=1;m<16;m<<=1){
    float ov = __shfl_xor(v, m);
    int   oi = __shfl_xor(i, m);
    if (ov > v || (ov == v && oi < i)) { v = ov; i = oi; }
  }
}
// merge per-lane online-softmax triples (max, sum, weighted-sum) across 16 lanes
static __device__ __forceinline__ void bmergestats16(float& m, float& s, float& e){
#pragma unroll
  for (int k=1;k<16;k<<=1){
    float om = __shfl_xor(m, k);
    float os = __shfl_xor(s, k);
    float oe = __shfl_xor(e, k);
    float nm = fmaxf(m, om);
    float sa = __expf(m - nm);
    float sb = __expf(om - nm);
    s = s*sa + os*sb;
    e = e*sa + oe*sb;
    m = nm;
  }
}

__global__ __launch_bounds__(256, 1)
void vq_fused(const float* __restrict__ z_e,
              const float* __restrict__ codebook,
              const float* __restrict__ gumbel,
              float* __restrict__ zq_out,
              float* __restrict__ hidx_out,
              float* __restrict__ kl_out,
              float* __restrict__ commit_out)
{
  __shared__ __align__(16) unsigned char smem[SMEM_TOT];
  __bf16* cbS  = (__bf16*)(smem + CB_OFF);
  __bf16* cbT  = (__bf16*)(smem + CBT_OFF);
  __bf16* zaA  = (__bf16*)(smem + ZA_OFF);
  float*  ttA  = (float*) (smem + TT_OFF);
  __bf16* wtA  = (__bf16*)(smem + WT_OFF);
  float*  csq  = (float*) (smem + CSQ_OFF);
  float*  zsq  = (float*) (smem + ZSQ_OFF);

  const int t    = threadIdx.x;
  const int wv   = t >> 5;
  const int lane = t & 31;
  const int h    = lane >> 4;   // half-wave select (kh)
  const int lc   = lane & 15;   // row (A) / col (B,C,D) index

  __bf16* za = zaA + wv * (RPW*ZA_STRIDE);
  float*  tt = ttA + wv * (RPW*TT_STRIDE);
  __bf16* wt = wtA + wv * (RPW*WT_STRIDE);

  const int r0 = blockIdx.x * RPB + wv * RPW;   // 16 rows: same b, contiguous w
  const int bb = r0 >> 11;
  const int w0 = r0 & 2047;

  // ---- load 16 z rows -> LDS bf16 (A-major), compute ||z||^2 per row ----
  {
    float acc = 0.f;
    const size_t baseZ = ((size_t)bb * FD) * WLEN + (size_t)w0 + lc;
#pragma unroll 4
    for (int j=0;j<128;++j){
      int f = j*2 + h;                                  // lanes cover even/odd feats
      float v = z_e[baseZ + (size_t)f * WLEN];          // coalesced over lc (w)
      acc += v*v;
      za[lc*ZA_STRIDE + f] = (__bf16)v;
    }
    acc += __shfl_xor(acc, 16);                         // pair covers all feats of row lc
    if (lane < 16) zsq[wv*RPW + lane] = acc;
  }
  float zs[8];
#pragma unroll
  for (int v=0;v<8;++v) zs[v] = zsq[wv*RPW + v + 8*h];

  const __bf16* zaRow = za + lc*ZA_STRIDE;
  const __bf16* wtRow = wt + lc*WT_STRIDE;

  // ---- state: index v covers row v (lanes 0-15) / row v+8 (lanes 16-31).
  // m1/am are lane-replicated (merged per tile); s1,m2,s2,e2 are PER-LANE
  // partials over the slots this lane owns, merged at the very end.
  float m1[8], s1[8], m2[8], s2[8], e2[8]; int am[8];
#pragma unroll
  for (int v=0;v<8;++v){ m1[v]=-3.0e38f; s1[v]=0.f; m2[v]=-3.0e38f; s2[v]=0.f; e2[v]=0.f; am[v]=0; }

  v8f accZ[16];                                         // z_q accumulators: 16 feat-subtiles
#pragma unroll
  for (int n=0;n<16;++n){ v8f zz = {}; accZ[n] = zz; }

  for (int tile=0; tile<NT; ++tile){
    __syncthreads();
    // cooperative codebook tile -> bf16, both [slot][feat] and [feat][slot]
    {
      const size_t gbase = (size_t)tile * TS * FD;
#pragma unroll 4
      for (int e = t; e < TS*FD; e += 256){
        int slot = e >> 8, f = e & 255;
        float c = codebook[gbase + (size_t)slot*FD + f];
        __bf16 cv = (__bf16)c;
        cbS[slot*CB_STRIDE + f]  = cv;
        cbT[f*CBT_STRIDE + slot] = cv;
      }
    }
    // prefetch next codebook tile into cache while this tile computes
    if (tile + 1 < NT){
      const char* np = (const char*)(codebook + (size_t)(tile+1) * TS * FD);
#pragma unroll
      for (int q=0;q<4;++q)
        __builtin_prefetch(np + ((size_t)t + (size_t)q*256) * 128, 0, 1);
    }
    __syncthreads();
    if (t < TS){
      float s = 0.f;
#pragma unroll 8
      for (int f=0; f<FD; ++f){ float c = (float)cbS[t*CB_STRIDE+f]; s += c*c; }
      csq[t] = s;
    }
    __syncthreads();

    // ---- sweep A: logits via WMMA, per-lane online stats, stash tempered logits ----
    float mtv[8]; int mti[8];                           // per-lane tile-local max/arg
#pragma unroll
    for (int v=0;v<8;++v){ mtv[v] = -3.0e38f; mti[v] = 0; }

    for (int sub=0; sub<8; ++sub){
      const int sc = sub*16;
      const __bf16* colbase = cbS + (sc+lc)*CB_STRIDE;
      v8f acc = {};
      // software-pipelined fragment double-buffering
      v16bf a0 = fragA(zaRow, 0, h);
      v16bf b0 = fragB(colbase, 0, h);
#pragma unroll
      for (int k=0;k<8;++k){                            // K over 256 feats
        v16bf a1 = (k < 7) ? fragA(zaRow, (k+1)*32, h) : a0;
        v16bf b1 = (k < 7) ? fragB(colbase, (k+1)*32, h) : b0;
        acc = __builtin_amdgcn_wmma_f32_16x16x32_bf16(false, a0, false, b0,
                                                      (short)0, acc, false, false);
        a0 = a1; b0 = b1;
      }
      const float cs = csq[sc + lc];
      const int slotG = tile*TS + sc + lc;
#pragma unroll
      for (int v=0; v<8; ++v){
        const int rowLoc = v + 8*h;
        float logit = 2.0f*acc[v] - zs[v] - cs;         // -distance
        float g  = gumbel[(size_t)(r0+rowLoc)*SLOTS + slotG];
        float tv = (logit + g) * 2.0f;                  // / tau (0.5)
        tt[rowLoc*TT_STRIDE + sc + lc] = tv;
        // per-lane online plain-softmax stats (no cross-lane traffic here)
        float nm = fmaxf(m2[v], logit);
        float sF = __expf(m2[v] - nm);
        float p  = __expf(logit - nm);
        s2[v] = s2[v]*sF + p;
        e2[v] = e2[v]*sF + logit*p;
        m2[v] = nm;
        // per-lane tempered tile-local max/argmax (compare only)
        if (tv > mtv[v]) { mtv[v] = tv; mti[v] = slotG; }
      }
    }

    // ---- merge tile max across lanes (once per tile); rescale if max moved ----
    float sc1[8]; bool need = false;
#pragma unroll
    for (int v=0;v<8;++v){
      bargmax16(mtv[v], mti[v]);                        // replicated per half-wave
      if (mtv[v] > m1[v]) am[v] = mti[v];
      float nm = fmaxf(m1[v], mtv[v]);
      sc1[v] = __expf(m1[v] - nm);
      m1[v]  = nm;
      need = need || (sc1[v] < 1.0f);
    }
    if (need){
#pragma unroll
      for (int v=0;v<8;++v) s1[v] *= sc1[v];            // per-lane partials scale linearly
#pragma unroll
      for (int n=0;n<16;++n){
#pragma unroll
        for (int v=0;v<8;++v) accZ[n][v] *= sc1[v];
      }
    }

    // ---- sweep B: w = exp(t - m1); per-lane normalizer partial; bf16 weights ----
#pragma unroll
    for (int sub=0; sub<8; ++sub){
      const int sc = sub*16;
#pragma unroll
      for (int v=0;v<8;++v){
        const int rowLoc = v + 8*h;
        float tv  = tt[rowLoc*TT_STRIDE + sc + lc];
        float wgt = __expf(tv - m1[v]);
        s1[v] += wgt;                                   // per-lane partial
        wt[rowLoc*WT_STRIDE + sc + lc] = (__bf16)wgt;
      }
    }

    // ---- z_q accumulation: accZ += w(16xTS) @ C_tile(TSxFD) ----
    {
      v16bf aw = fragA(wtRow, 0, h);
#pragma unroll
      for (int k=0;k<4;++k){                            // K over 128 slots
        v16bf awn = (k < 3) ? fragA(wtRow, (k+1)*32, h) : aw;
        v16bf bcur = fragB(cbT + (0*16+lc)*CBT_STRIDE, k*32, h);
#pragma unroll
        for (int n=0;n<16;++n){
          v16bf bnext = (n < 15) ? fragB(cbT + ((n+1)*16+lc)*CBT_STRIDE, k*32, h)
                                 : bcur;
          accZ[n] = __builtin_amdgcn_wmma_f32_16x16x32_bf16(false, aw, false, bcur,
                                                            (short)0, accZ[n], false, false);
          bcur = bnext;
        }
        aw = awn;
      }
    }
  }

  // ---- final cross-lane merges (once) ----
  float inv1[8];
#pragma unroll
  for (int v=0;v<8;++v){
    float s1t = bsum16(s1[v]);                          // total tempered normalizer
    inv1[v] = 1.0f / s1t;
    bmergestats16(m2[v], s2[v], e2[v]);                 // plain-softmax stats
  }

  // ---- normalize + transpose-store z_q ----
  float* tb = tt;                                       // reuse wave-private region, stride 17
#pragma unroll
  for (int n=0;n<16;++n){
#pragma unroll
    for (int v=0;v<8;++v)
      tb[(v + 8*h)*17 + lc] = accZ[n][v] * inv1[v];
    // same-wave LDS is in-order: loads below see the stores
#pragma unroll
    for (int j=0;j<8;++j){
      int fi = j*2 + h;
      float val = tb[lc*17 + fi];                       // (row=w-index lc, feat fi)
      zq_out[((size_t)(bb*FD + n*16 + fi))*WLEN + w0 + lc] = val;  // coalesced in w
    }
  }

#pragma unroll
  for (int v=0;v<8;++v)
    if (lc == 0) hidx_out[r0 + v + 8*h] = (float)am[v];

  float kls = 0.f, cms = 0.f;
#pragma unroll
  for (int v=0;v<8;++v){
    float eb  = e2[v] / s2[v];                          // E_probs[logits]
    float lse = m2[v] + __logf(s2[v]);
    kls += eb - lse + 6.93147180559945f;                // + log(1024)
    cms += -eb;                                         // sum p*d
  }
  kls += __shfl_xor(kls, 16);
  cms += __shfl_xor(cms, 16);
  if (lane == 0){
    atomicAdd(kl_out,     kls * (1.0f/32.0f));
    atomicAdd(commit_out, cms * (1.0f/32.0f));
  }
}

__global__ void vq_init_scalars(float* kl, float* commit){
  if (threadIdx.x == 0){ *kl = 0.f; *commit = 0.f; }
}

extern "C" void kernel_launch(void* const* d_in, const int* in_sizes, int n_in,
                              void* d_out, int out_size, void* d_ws, size_t ws_size,
                              hipStream_t stream) {
  const float* z_e  = (const float*)d_in[0];
  const float* cbk  = (const float*)d_in[1];
  const float* gum  = (const float*)d_in[2];
  float* zq     = (float*)d_out;
  float* hidx   = zq + (size_t)BSZ*FD*WLEN;   // 16,777,216
  float* kl     = hidx + NROWS;               // +65,536
  float* commit = kl + 1;

  hipLaunchKernelGGL(vq_init_scalars, dim3(1), dim3(64), 0, stream, kl, commit);
  hipLaunchKernelGGL(vq_fused, dim3(NROWS/RPB), dim3(256), 0, stream,
                     z_e, cbk, gum, zq, hidx, kl, commit);
}
